// SAGE_12541304504870
// MI455X (gfx1250) — matrix operand order repaired and checked by reading, not compiled, not run on hardware
//
#include <hip/hip_runtime.h>
#include <hip/hip_bf16.h>
#include <stdint.h>

// ---------------------------------------------------------------------------
// GraphSAGE 3-layer inference for gfx1250 (MI455X), wave32 + WMMA bf16.
//   layer: out = relu?( mean_agg(h)[.]@Wl + h@Wr + bl )
// Aggregation via fp32 global atomics (20MB node matrix lives in 192MB L2).
// GEMM via v_wmma_f32_16x16x32_bf16, one 16x16 output tile per wave.
// ---------------------------------------------------------------------------

typedef __attribute__((ext_vector_type(16))) __bf16 v16bf;
typedef __attribute__((ext_vector_type(8)))  float  v8f;

#define FDIM 128   // K width of every GEMM (F_IN == HID == 128)

// ---------------------------- utility kernels ------------------------------

__global__ void sage_fill_zero(float* __restrict__ p, int n) {
  int i = blockIdx.x * blockDim.x + threadIdx.x;
  if (i < n) p[i] = 0.0f;
}

__global__ void sage_degree(const long long* __restrict__ dst,
                            float* __restrict__ deg, int E) {
  int e = blockIdx.x * blockDim.x + threadIdx.x;
  if (e < E) atomicAdd(&deg[(int)dst[e]], 1.0f);
}

__global__ void sage_invdeg(float* __restrict__ deg, int n) {
  int i = blockIdx.x * blockDim.x + threadIdx.x;
  if (i < n) deg[i] = 1.0f / fmaxf(deg[i], 1.0f);
}

// Pack fp32 weight W[din, dout] (row major) into the wave32 WMMA B-fragment
// layout for v_wmma_f32_16x16x32_bf16:
//   per (ktile, ntile): 32 lanes x 16 bf16 (32B) contiguous.
//   lanes 0-15: N = ntile*16+lane,    K = ktile*32 + 0..7, 16..23
//   lanes16-31: N = ntile*16+lane-16, K = ktile*32 + 8..15, 24..31
__global__ void sage_pack_w(const float* __restrict__ W, __bf16* __restrict__ Wp,
                            int dout, int ktiles, int ntiles) {
  int t = blockIdx.x * blockDim.x + threadIdx.x;
  int total = ktiles * ntiles * 32;
  if (t >= total) return;
  int lane = t & 31;
  int tile = t >> 5;
  int nt = tile % ntiles;
  int kt = tile / ntiles;
  int n = nt * 16 + (lane & 15);
  int kbase = kt * 32 + (lane >> 4) * 8;
  __bf16* o = Wp + (size_t)t * 16;
#pragma unroll
  for (int i = 0; i < 16; ++i)
    o[i] = (__bf16)W[(size_t)(kbase + i + (i >= 8 ? 8 : 0)) * dout + n];
}

// Scatter-add: agg[dst] += h[src], 4 features per thread (float4 gather).
__global__ void sage_aggregate(const float* __restrict__ h,
                               const long long* __restrict__ src,
                               const long long* __restrict__ dst,
                               float* __restrict__ agg, int E) {
  int t = blockIdx.x * blockDim.x + threadIdx.x;
  int e = t >> 5;                 // 32 chunks of 4 floats = 128 features
  if (e >= E) return;
  int c = (t & 31) << 2;
  long long s = src[e], d = dst[e];
  const float4 v = *(const float4*)(h + (size_t)s * FDIM + c);
  float* o = agg + (size_t)d * FDIM + c;
  atomicAdd(o + 0, v.x);
  atomicAdd(o + 1, v.y);
  atomicAdd(o + 2, v.z);
  atomicAdd(o + 3, v.w);
}

// ------------------------------ WMMA GEMM ----------------------------------

// Build an A-fragment (16x32 bf16, wave32 layout) from a row-major fp32 row,
// scaled by s.  Per the ISA 16-bit A layout:
//   lanes 0-15 (M=lane):    elems 0..7 = K 0..7,  elems 8..15 = K 16..23
//   lanes16-31 (M=lane-16): elems 0..7 = K 8..15, elems 8..15 = K 24..31
// `p` must already point at row + (lane>>4)*8.
__device__ inline v16bf sage_load_a(const float* __restrict__ p, float s) {
  float4 x0 = *(const float4*)(p + 0);
  float4 x1 = *(const float4*)(p + 4);
  float4 x2 = *(const float4*)(p + 16);
  float4 x3 = *(const float4*)(p + 20);
  v16bf a;
  a[0]  = (__bf16)(x0.x * s); a[1]  = (__bf16)(x0.y * s);
  a[2]  = (__bf16)(x0.z * s); a[3]  = (__bf16)(x0.w * s);
  a[4]  = (__bf16)(x1.x * s); a[5]  = (__bf16)(x1.y * s);
  a[6]  = (__bf16)(x1.z * s); a[7]  = (__bf16)(x1.w * s);
  a[8]  = (__bf16)(x2.x * s); a[9]  = (__bf16)(x2.y * s);
  a[10] = (__bf16)(x2.z * s); a[11] = (__bf16)(x2.w * s);
  a[12] = (__bf16)(x3.x * s); a[13] = (__bf16)(x3.y * s);
  a[14] = (__bf16)(x3.z * s); a[15] = (__bf16)(x3.w * s);
  return a;
}

// out[mt*16.., nt*16..] = (invdeg[m]*A) @ Wl + H @ Wr + bias, optional relu.
// block = (32, ntiles) -> one wave per 16x16 output tile; grid.x = N/16.
__global__ void sage_gemm(const float* __restrict__ A,   // agg_sum [N,128]
                          const float* __restrict__ H,   // root features [N,128]
                          const float* __restrict__ invdeg,
                          const v16bf* __restrict__ WlP,
                          const v16bf* __restrict__ WrP,
                          const float* __restrict__ bias,
                          float* __restrict__ out,
                          int dout, int ntiles, int do_relu) {
  const int lane = threadIdx.x;          // 0..31, one wave per threadIdx.y
  const int nt   = threadIdx.y;          // output column tile
  const int mt   = blockIdx.x;           // output row tile
  const int m    = mt * 16 + (lane & 15);
  const int kh   = (lane >> 4) * 8;      // K sub-offset per A layout

  v8f acc = {};
  const float s = invdeg[m];
  const float* arow = A + (size_t)m * FDIM + kh;
  const float* hrow = H + (size_t)m * FDIM + kh;

#pragma unroll
  for (int kt = 0; kt < FDIM / 32; ++kt) {           // agg @ Wl
    v16bf a = sage_load_a(arow + kt * 32, s);
    v16bf b = WlP[(size_t)(kt * ntiles + nt) * 32 + lane];
    acc = __builtin_amdgcn_wmma_f32_16x16x32_bf16(
        false, a, false, b, (short)0, acc, false, false);
  }
#pragma unroll
  for (int kt = 0; kt < FDIM / 32; ++kt) {           // h @ Wr
    v16bf a = sage_load_a(hrow + kt * 32, 1.0f);
    v16bf b = WrP[(size_t)(kt * ntiles + nt) * 32 + lane];
    acc = __builtin_amdgcn_wmma_f32_16x16x32_bf16(
        false, a, false, b, (short)0, acc, false, false);
  }

  // C/D layout: VGPR r -> row (lane<16 ? r : r+8), col = lane&15.
  const int col = nt * 16 + (lane & 15);
  const float bv = bias[col];
  float* orow = out + ((size_t)mt * 16 + kh) * dout + col;
#pragma unroll
  for (int r = 0; r < 8; ++r) {
    float v = acc[r] + bv;
    if (do_relu) v = fmaxf(v, 0.0f);
    orow[(size_t)r * dout] = v;
  }
}

// ------------------------------- launcher ----------------------------------

extern "C" void kernel_launch(void* const* d_in, const int* in_sizes, int n_in,
                              void* d_out, int out_size, void* d_ws, size_t ws_size,
                              hipStream_t stream) {
  const float*     x    = (const float*)d_in[0];
  const long long* ei   = (const long long*)d_in[1];
  const float* Wl0 = (const float*)d_in[2];
  const float* bl0 = (const float*)d_in[3];
  const float* Wr0 = (const float*)d_in[4];
  const float* Wl1 = (const float*)d_in[5];
  const float* bl1 = (const float*)d_in[6];
  const float* Wr1 = (const float*)d_in[7];
  const float* Wl2 = (const float*)d_in[8];
  const float* bl2 = (const float*)d_in[9];
  const float* Wr2 = (const float*)d_in[10];

  const int N    = in_sizes[0] / FDIM;   // 40000
  const int E    = in_sizes[1] / 2;      // 640000
  const int NCLS = in_sizes[9];          // 64 (size of bl2)

  const long long* src = ei;
  const long long* dst = ei + E;

  // ---- workspace carve-out (256B aligned) ----
  char* w = (char*)d_ws;
  auto carve = [&](size_t bytes) -> void* {
    void* p = (void*)w;
    w += (bytes + 255) & ~(size_t)255;
    return p;
  };
  float*  deg  = (float*)carve((size_t)N * 4);
  float*  agg  = (float*)carve((size_t)N * FDIM * 4);
  float*  h1   = (float*)carve((size_t)N * FDIM * 4);
  float*  h2   = (float*)carve((size_t)N * FDIM * 4);
  __bf16* Wl0p = (__bf16*)carve((size_t)FDIM * FDIM * 2);
  __bf16* Wr0p = (__bf16*)carve((size_t)FDIM * FDIM * 2);
  __bf16* Wl1p = (__bf16*)carve((size_t)FDIM * FDIM * 2);
  __bf16* Wr1p = (__bf16*)carve((size_t)FDIM * FDIM * 2);
  __bf16* Wl2p = (__bf16*)carve((size_t)FDIM * NCLS * 2);
  __bf16* Wr2p = (__bf16*)carve((size_t)FDIM * NCLS * 2);
  (void)ws_size; (void)n_in; (void)out_size;

  const int T = 256;
  const int ktiles = FDIM / 32;          // 4
  const int ntH    = FDIM / 16;          // 8 column tiles for hidden layers
  const int ntO    = NCLS / 16;          // 4 column tiles for output layer

  // ---- degree -> reciprocal (once) ----
  sage_fill_zero<<<(N + T - 1) / T, T, 0, stream>>>(deg, N);
  sage_degree   <<<(E + T - 1) / T, T, 0, stream>>>(dst, deg, E);
  sage_invdeg   <<<(N + T - 1) / T, T, 0, stream>>>(deg, N);

  // ---- pack all weights into WMMA B-fragment layout (once) ----
  sage_pack_w<<<(ktiles * ntH * 32 + T - 1) / T, T, 0, stream>>>(Wl0, Wl0p, FDIM, ktiles, ntH);
  sage_pack_w<<<(ktiles * ntH * 32 + T - 1) / T, T, 0, stream>>>(Wr0, Wr0p, FDIM, ktiles, ntH);
  sage_pack_w<<<(ktiles * ntH * 32 + T - 1) / T, T, 0, stream>>>(Wl1, Wl1p, FDIM, ktiles, ntH);
  sage_pack_w<<<(ktiles * ntH * 32 + T - 1) / T, T, 0, stream>>>(Wr1, Wr1p, FDIM, ktiles, ntH);
  sage_pack_w<<<(ktiles * ntO * 32 + T - 1) / T, T, 0, stream>>>(Wl2, Wl2p, NCLS, ktiles, ntO);
  sage_pack_w<<<(ktiles * ntO * 32 + T - 1) / T, T, 0, stream>>>(Wr2, Wr2p, NCLS, ktiles, ntO);

  const int nAgg   = N * FDIM;
  const int aggThr = E * 32;             // (edge, 4-feature chunk) threads
  const dim3 gemmGridH((unsigned)(N / 16));
  const dim3 gemmBlkH(32, (unsigned)ntH);
  const dim3 gemmBlkO(32, (unsigned)ntO);

  // ---- layer 0: x -> h1 (relu) ----
  sage_fill_zero<<<(nAgg + T - 1) / T, T, 0, stream>>>(agg, nAgg);
  sage_aggregate<<<(aggThr + T - 1) / T, T, 0, stream>>>(x, src, dst, agg, E);
  sage_gemm<<<gemmGridH, gemmBlkH, 0, stream>>>(
      agg, x, deg, (const v16bf*)Wl0p, (const v16bf*)Wr0p, bl0, h1, FDIM, ntH, 1);

  // ---- layer 1: h1 -> h2 (relu) ----
  sage_fill_zero<<<(nAgg + T - 1) / T, T, 0, stream>>>(agg, nAgg);
  sage_aggregate<<<(aggThr + T - 1) / T, T, 0, stream>>>(h1, src, dst, agg, E);
  sage_gemm<<<gemmGridH, gemmBlkH, 0, stream>>>(
      agg, h1, deg, (const v16bf*)Wl1p, (const v16bf*)Wr1p, bl1, h2, FDIM, ntH, 1);

  // ---- layer 2: h2 -> out (no relu) ----
  sage_fill_zero<<<(nAgg + T - 1) / T, T, 0, stream>>>(agg, nAgg);
  sage_aggregate<<<(aggThr + T - 1) / T, T, 0, stream>>>(h2, src, dst, agg, E);
  sage_gemm<<<gemmGridH, gemmBlkO, 0, stream>>>(
      agg, h2, deg, (const v16bf*)Wl2p, (const v16bf*)Wr2p, bl2, (float*)d_out,
      NCLS, ntO, 0);
}
